// GlimpseNet_35244501631036
// MI455X (gfx1250) — matrix operand order repaired and checked by reading, not compiled, not run on hardware
//
#include <hip/hip_runtime.h>
#include <cstdint>

// ---------------- problem constants (from reference) ----------------
#define B_SZ   512
#define CCH    3
#define IMG    256
#define D_IN   9216     // NPATCH * PATCH * PATCH * C = 3*32*32*3
#define HG_SZ  1024
#define HL_SZ  128
#define NOUT   1152     // HG + HL

typedef __attribute__((ext_vector_type(16))) __bf16 v16bf;
typedef __attribute__((ext_vector_type(8)))  float  v8f;

struct U8 { uint32_t x[8]; };

#define LDS_ROW    80                 // bytes per 32-bf16 row (64B data + 16B pad: bank-conflict-free)
#define TILE_BYTES (64 * LDS_ROW)     // one 64-row tile = 5120 B

// f32 -> bf16 round-to-nearest-even
static __device__ __forceinline__ unsigned short f2bf(float f) {
  uint32_t u = __builtin_bit_cast(uint32_t, f);
  u += 0x7fffu + ((u >> 16) & 1u);
  return (unsigned short)(u >> 16);
}

// ---------------- foveation: gather + avg-pool, emit bf16 glimpse ----------------
__global__ void k_foveate(const float* __restrict__ x, const float* __restrict__ l,
                          unsigned short* __restrict__ g) {
  int idx = blockIdx.x * blockDim.x + threadIdx.x;     // B_SZ * D_IN threads
  int b   = idx / D_IN;
  int r   = idx % D_IN;
  int ch  = r >> 10;          // 0..8
  int pix = r & 1023;
  int s_i = ch / 3;           // scale index
  int c   = ch % 3;
  int pi  = pix >> 5, pj = pix & 31;

  float lx = l[b * 2 + 0], ly = l[b * 2 + 1];
  int fx = (int)floorf(0.5f * ((lx + 1.0f) * 256.0f));
  int fy = (int)floorf(0.5f * ((ly + 1.0f) * 256.0f));
  fx = min(max(fx, 0), IMG);
  fy = min(max(fy, 0), IMG);

  int size = 32 << s_i;
  int pad  = size >> 1;
  int kds  = 1 << s_i;        // pooling factor

  const float* xc = x + ((size_t)b * CCH + c) * (IMG * IMG);
  float sum = 0.0f;
  for (int di = 0; di < kds; ++di) {
    int yy = fy + pi * kds + di - pad;
    if (yy < 0 || yy >= IMG) continue;
    for (int dj = 0; dj < kds; ++dj) {
      int xx = fx + pj * kds + dj - pad;
      if (xx < 0 || xx >= IMG) continue;
      sum += xc[yy * IMG + xx];
    }
  }
  g[idx] = f2bf(sum * (1.0f / (float)(kds * kds)));
}

// ---------------- generic f32 -> bf16 conversion ----------------
__global__ void k_cvt(const float* __restrict__ src, unsigned short* __restrict__ dst, int n) {
  for (int i = blockIdx.x * blockDim.x + threadIdx.x; i < n; i += gridDim.x * blockDim.x)
    dst[i] = f2bf(src[i]);
}

// ---------------- hw = relu(l_t @ W2.T + b2), K=2 ----------------
__global__ void k_hw(const float* __restrict__ l, const float* __restrict__ W2,
                     const float* __restrict__ b2, unsigned short* __restrict__ hwb) {
  int idx = blockIdx.x * blockDim.x + threadIdx.x;     // B_SZ * HL_SZ threads
  int b = idx >> 7, n = idx & 127;
  float v = l[b * 2 + 0] * W2[n * 2 + 0] + l[b * 2 + 1] * W2[n * 2 + 1] + b2[n];
  hwb[idx] = f2bf(fmaxf(v, 0.0f));
}

// ---------------- CDNA5 async copy: 16B global -> LDS, tracked by ASYNCcnt ----------------
// global_load_async_to_lds_b128: VDST = per-lane LDS byte offset, VADDR = 64-bit global addr
static __device__ __forceinline__ void async_cp16(uint32_t ldsOff, const void* g) {
  asm volatile("global_load_async_to_lds_b128 %0, %1, off"
               :: "v"(ldsOff), "v"((unsigned long long)(uintptr_t)g)
               : "memory");
}
static __device__ __forceinline__ void wait_async0() {
  asm volatile("s_wait_asynccnt 0x0" ::: "memory");
}
// Same wait, but with fake data deps on the accumulators so the scheduler cannot
// hoist it above the WMMAs: keeps async DMA (next stage) overlapped with compute.
static __device__ __forceinline__ void wait_async0_after(float d0, float d1) {
  asm volatile("s_wait_asynccnt 0x0" :: "v"(d0), "v"(d1) : "memory");
}

// stage one 64-row x 32-col bf16 tile (row-major, row length K) into LDS (row stride LDS_ROW)
// 256 threads: thread = row*4 + 16B-chunk -> one async b128 each
static __device__ __forceinline__ void stage_tile(const unsigned short* __restrict__ src,
                                                  int row0, int K, int k0,
                                                  uint32_t ldsBaseOff, int tid) {
  int row = tid >> 2;
  int ch  = tid & 3;
  const void* g = (const uint8_t*)(src + (size_t)(row0 + row) * K + k0) + ch * 16;
  async_cp16(ldsBaseOff + (uint32_t)(row * LDS_ROW + ch * 16), g);
}

// ---------------- WMMA fragment loads from LDS (layouts per ISA 7.12.2) ----------------
static __device__ __forceinline__ v16bf load_afrag(const uint8_t* bufA, int wm, int l15, int half) {
  const uint8_t* p = bufA + (wm * 16 + l15) * LDS_ROW + half * 16;
  uint4 a0 = *(const uint4*)p;           // A: K = (v/4)*16 + h*8 + (v%4)*2
  uint4 a1 = *(const uint4*)(p + 32);
  U8 u = {{a0.x, a0.y, a0.z, a0.w, a1.x, a1.y, a1.z, a1.w}};
  return __builtin_bit_cast(v16bf, u);
}
static __device__ __forceinline__ v16bf load_bfrag(const uint8_t* bufB, int nloc, int half) {
  const uint8_t* p = bufB + nloc * LDS_ROW + half * 16;
  uint4 b0 = *(const uint4*)p;           // B: K = (v/4)*16 + h*8 + (v%4)*2
  uint4 b1 = *(const uint4*)(p + 32);
  U8 u = {{b0.x, b0.y, b0.z, b0.w, b1.x, b1.y, b1.z, b1.w}};
  return __builtin_bit_cast(v16bf, u);
}

// ---------------- double-buffered, async-staged WMMA K-loop ----------------
// Block tile 64(M) x 64(N); wave (wm = w&3, wn = w>>2) computes 16x32 via 2 accumulators.
static __device__ __forceinline__ void gemm_phase(
    const unsigned short* __restrict__ A, int arow0,
    const unsigned short* __restrict__ Bw, int brow0, int K,
    uint8_t* smem, uint32_t smemOff, v8f acc[2]) {
  const int tid  = threadIdx.x;
  const int lane = tid & 31, wv = tid >> 5;
  const int wm = wv & 3, wn = wv >> 2;
  const int half = lane >> 4, l15 = lane & 15;

  // smem layout: [buf(2)][A=0/B=1][TILE_BYTES]
  stage_tile(A,  arow0, K, 0, smemOff + 0 * TILE_BYTES, tid);
  stage_tile(Bw, brow0, K, 0, smemOff + 1 * TILE_BYTES, tid);
  wait_async0();
  __syncthreads();

  int cur = 0;
  for (int k0 = 0; k0 < K; k0 += 32) {
    int nxt = cur ^ 1;
    if (k0 + 32 < K) {   // prefetch next stage while computing current
      stage_tile(A,  arow0, K, k0 + 32, smemOff + (uint32_t)((nxt * 2 + 0) * TILE_BYTES), tid);
      stage_tile(Bw, brow0, K, k0 + 32, smemOff + (uint32_t)((nxt * 2 + 1) * TILE_BYTES), tid);
    }
    const uint8_t* bufA = smem + (cur * 2 + 0) * TILE_BYTES;
    const uint8_t* bufB = smem + (cur * 2 + 1) * TILE_BYTES;
    v16bf af = load_afrag(bufA, wm, l15, half);
#pragma unroll
    for (int t = 0; t < 2; ++t) {
      v16bf bf = load_bfrag(bufB, wn * 32 + t * 16 + l15, half);
      acc[t] = __builtin_amdgcn_wmma_f32_16x16x32_bf16(
          false, af, false, bf, (short)0, acc[t], false, false);
    }
    wait_async0_after(acc[0][0], acc[1][0]);  // next-stage copies done; pinned after WMMAs
    __syncthreads();                          // visible to all waves; all reads of cur done
    cur = nxt;
  }
}

// ---------------- GEMM1: h = relu(glimpse @ W1.T + b1), bf16 out ----------------
__global__ __launch_bounds__(256) void k_gemm1(
    const unsigned short* __restrict__ Ag, const unsigned short* __restrict__ Bw,
    const float* __restrict__ bias, unsigned short* __restrict__ outb) {
  __shared__ __align__(16) uint8_t smem[2 * 2 * TILE_BYTES];
  const uint32_t smemOff = (uint32_t)(uintptr_t)&smem[0];   // LDS aperture: low 32 bits = LDS byte addr
  const int lane = threadIdx.x & 31, wv = threadIdx.x >> 5;
  const int wm = wv & 3, wn = wv >> 2;
  const int half = lane >> 4, l15 = lane & 15;
  const int N = HG_SZ;

  v8f acc[2] = {};
  gemm_phase(Ag, blockIdx.x * 64, Bw, blockIdx.y * 64, D_IN, smem, smemOff, acc);

#pragma unroll
  for (int t = 0; t < 2; ++t) {
    int ncol = blockIdx.y * 64 + wn * 32 + t * 16 + l15;
    float bs = bias[ncol];
#pragma unroll
    for (int r = 0; r < 8; ++r) {                 // D: m = r + 8*half, n = lane&15
      int mm = blockIdx.x * 64 + wm * 16 + r + (half << 3);
      outb[(size_t)mm * N + ncol] = f2bf(fmaxf(acc[t][r] + bs, 0.0f));
    }
  }
}

// ---------------- GEMM2 (fused): out = relu(h@W3.T + b3 + hw@W4.T + b4), f32 out ----------------
__global__ __launch_bounds__(256) void k_gemm2(
    const unsigned short* __restrict__ Ah, const unsigned short* __restrict__ W3b,
    const float* __restrict__ b3,
    const unsigned short* __restrict__ Aw, const unsigned short* __restrict__ W4b,
    const float* __restrict__ b4, float* __restrict__ out) {
  __shared__ __align__(16) uint8_t smem[2 * 2 * TILE_BYTES];
  const uint32_t smemOff = (uint32_t)(uintptr_t)&smem[0];
  const int lane = threadIdx.x & 31, wv = threadIdx.x >> 5;
  const int wm = wv & 3, wn = wv >> 2;
  const int half = lane >> 4, l15 = lane & 15;
  const int N = NOUT;

  v8f acc[2] = {};
  gemm_phase(Ah, blockIdx.x * 64, W3b, blockIdx.y * 64, HG_SZ, smem, smemOff, acc); // "what"
  gemm_phase(Aw, blockIdx.x * 64, W4b, blockIdx.y * 64, HL_SZ, smem, smemOff, acc); // + "where"

#pragma unroll
  for (int t = 0; t < 2; ++t) {
    int ncol = blockIdx.y * 64 + wn * 32 + t * 16 + l15;
    float bs = b3[ncol] + b4[ncol];
#pragma unroll
    for (int r = 0; r < 8; ++r) {
      int mm = blockIdx.x * 64 + wm * 16 + r + (half << 3);
      out[(size_t)mm * N + ncol] = fmaxf(acc[t][r] + bs, 0.0f);
    }
  }
}

// ---------------- launcher ----------------
extern "C" void kernel_launch(void* const* d_in, const int* in_sizes, int n_in,
                              void* d_out, int out_size, void* d_ws, size_t ws_size,
                              hipStream_t stream) {
  const float* x_t = (const float*)d_in[0];
  const float* l_t = (const float*)d_in[1];
  const float* W1  = (const float*)d_in[2];
  const float* b1  = (const float*)d_in[3];
  const float* W2  = (const float*)d_in[4];
  const float* b2  = (const float*)d_in[5];
  const float* W3  = (const float*)d_in[6];
  const float* b3  = (const float*)d_in[7];
  const float* W4  = (const float*)d_in[8];
  const float* b4  = (const float*)d_in[9];
  float* out = (float*)d_out;

  char* ws = (char*)d_ws;
  size_t off = 0;
  auto wsalloc = [&](size_t bytes) -> void* {
    void* p = ws + off;
    off = (off + bytes + 255) & ~(size_t)255;
    return p;
  };
  unsigned short* g_b  = (unsigned short*)wsalloc((size_t)B_SZ * D_IN * 2);   //  9.4 MB glimpse bf16
  unsigned short* W1b  = (unsigned short*)wsalloc((size_t)HG_SZ * D_IN * 2);  // 18.9 MB
  unsigned short* W3bb = (unsigned short*)wsalloc((size_t)NOUT * HG_SZ * 2);  //  2.4 MB
  unsigned short* W4bb = (unsigned short*)wsalloc((size_t)NOUT * HL_SZ * 2);  //  0.3 MB
  unsigned short* h_b  = (unsigned short*)wsalloc((size_t)B_SZ * HG_SZ * 2);  //  1.0 MB
  unsigned short* hw_b = (unsigned short*)wsalloc((size_t)B_SZ * HL_SZ * 2);  //  0.13 MB

  k_foveate<<<(B_SZ * D_IN) / 256, 256, 0, stream>>>(x_t, l_t, g_b);
  k_cvt<<<2048, 256, 0, stream>>>(W1, W1b, HG_SZ * D_IN);
  k_cvt<<<1024, 256, 0, stream>>>(W3, W3bb, NOUT * HG_SZ);
  k_cvt<<<256,  256, 0, stream>>>(W4, W4bb, NOUT * HL_SZ);
  k_hw<<<(B_SZ * HL_SZ) / 256, 256, 0, stream>>>(l_t, W2, b2, hw_b);
  k_gemm1<<<dim3(B_SZ / 64, HG_SZ / 64), 256, 0, stream>>>(g_b, W1b, b1, h_b);
  k_gemm2<<<dim3(B_SZ / 64, NOUT / 64), 256, 0, stream>>>(h_b, W3bb, b3, hw_b, W4bb, b4, out);
}
